// HMLSTMOutput_83631603188456
// MI455X (gfx1250) — matrix operand order repaired and checked by reading, not compiled, not run on hardware
//
#include <hip/hip_runtime.h>

// ---------------- problem constants ----------------
#define B_   4
#define T_   1024
#define L_   3
#define DIN_ 1024
#define D_   3072          // L_*DIN_
#define EMB_ 2048
#define OUT_ 32000
#define M_   (B_*T_)       // 4096 tokens

// ---------------- workspace layout (bytes) ----------------
// all offsets multiples of 2048 -> 16B alignment guaranteed
#define OFF_XS    ((size_t)0)                       // bf16 [4096][3072]   25,165,824
#define OFF_EMBT  ((size_t)25165824)                // bf16 [2048][3072]   12,582,912
#define OFF_LINT0 ((size_t)37748736)                // bf16 [2048][2048]    8,388,608
#define OFF_LINT1 ((size_t)46137344)                // bf16 [2048][2048]    8,388,608
#define OFF_OUTT  ((size_t)54525952)                // bf16 [32000][2048] 131,072,000
#define OFF_EBIAS ((size_t)185597952)               // f32  [2048]              8,192
#define OFF_H0    ((size_t)185606144)               // bf16 [4096][2048]   16,777,216
#define OFF_H1    ((size_t)202383360)               // bf16 [4096][2048]   16,777,216

typedef __attribute__((ext_vector_type(16))) __bf16 v16bf;
typedef __attribute__((ext_vector_type(8)))  float  v8f;

__device__ __forceinline__ unsigned short f2bf(float f) {
  unsigned int u = __builtin_bit_cast(unsigned int, f);
  u += 0x7FFFu + ((u >> 16) & 1u);                 // round-to-nearest-even
  return (unsigned short)(u >> 16);
}

// ---------------- kernel: emb_b summed over layers ----------------
__global__ void ebias_kernel(const float* __restrict__ eb, float* __restrict__ out) {
  int e = blockIdx.x * 256 + threadIdx.x;
  out[e] = eb[e] + eb[EMB_ + e] + eb[2 * EMB_ + e];
}

// ---------------- kernel: per-token gates + gate-scaled x in bf16 ----------------
__global__ __launch_bounds__(256) void gates_kernel(const float* __restrict__ x,
                                                    const float* __restrict__ w,
                                                    unsigned short* __restrict__ xs) {
  const int t   = blockIdx.x;
  const int tid = threadIdx.x;
  const float* xr = x + (size_t)t * D_;
  __shared__ float red[3 * 256];
  __shared__ float gsh[3];

  float a0 = 0.f, a1 = 0.f, a2 = 0.f;
#pragma unroll
  for (int j = 0; j < 12; ++j) {
    int d = j * 256 + tid;
    float xv = xr[d];
    a0 += xv * w[d];
    a1 += xv * w[D_ + d];
    a2 += xv * w[2 * D_ + d];
  }
  red[tid] = a0; red[256 + tid] = a1; red[512 + tid] = a2;
  __syncthreads();
  for (int s = 128; s > 0; s >>= 1) {
    if (tid < s) {
      red[tid]       += red[tid + s];
      red[256 + tid] += red[256 + tid + s];
      red[512 + tid] += red[512 + tid + s];
    }
    __syncthreads();
  }
  if (tid < 3) gsh[tid] = 1.f / (1.f + __expf(-red[tid * 256]));
  __syncthreads();

  unsigned short* yr = xs + (size_t)t * D_;
#pragma unroll
  for (int j = 0; j < 12; ++j) {
    int d = j * 256 + tid;
    yr[d] = f2bf(xr[d] * gsh[d >> 10]);           // d>>10 = layer index
  }
}

// ---------------- kernel: fp32 [R][C] -> bf16 [C][R] (convert + transpose) ----------------
__global__ __launch_bounds__(256) void convT_kernel(const float* __restrict__ src,
                                                    unsigned short* __restrict__ dst,
                                                    int R, int C) {
  __shared__ unsigned short tile[32][33];
  const int c0 = blockIdx.x * 32, r0 = blockIdx.y * 32;
  const int tx = threadIdx.x, ty = threadIdx.y;   // 32 x 8
#pragma unroll
  for (int i = 0; i < 4; ++i) {
    int r = r0 + ty + i * 8;
    tile[ty + i * 8][tx] = f2bf(src[(size_t)r * C + c0 + tx]);
  }
  __syncthreads();
#pragma unroll
  for (int i = 0; i < 4; ++i) {
    int c = c0 + ty + i * 8;
    dst[(size_t)c * R + r0 + tx] = tile[tx][ty + i * 8];
  }
}

// ---------------- WMMA GEMM: C = act(A[M][K] @ Bt[N][K]^T + bias) ----------------
// ACT: 0 = ReLU -> bf16, 1 = tanh -> bf16, 2 = none -> fp32
// 128x128 tile per 256-thread block, 8 waves in 2x4, each wave 4x2 16x16 accum tiles.
// Double-buffered LDS with register-staged global prefetch (tile i+2 in flight
// while tile i computes) + global_prefetch of tile i+3.
template <int ACT>
__global__ __launch_bounds__(256) void gemm_kernel(const unsigned short* __restrict__ A,
                                                   const unsigned short* __restrict__ Bt,
                                                   const float* __restrict__ bias,
                                                   void* __restrict__ Cout,
                                                   int Ndim, int K) {
  constexpr int LDT = 40;                         // padded LDS row stride (bf16 units)
  __shared__ unsigned short As[2][128 * LDT];
  __shared__ unsigned short Bs[2][128 * LDT];

  const int tid  = threadIdx.x;
  const int lane = tid & 31;
  const int wave = tid >> 5;                      // 0..7
  const int wm   = wave >> 2;                     // 0..1 -> 64 rows each
  const int wn   = wave & 3;                      // 0..3 -> 32 cols each
  const int bm   = blockIdx.y * 128;
  const int bn   = blockIdx.x * 128;

  const int half = lane >> 4;                     // 0: lanes 0-15, 1: lanes 16-31
  const int lr   = lane & 15;

  // staging geometry: 512 16B chunks per tile, 2 per thread
  const int ch0 = tid, ch1 = tid + 256;
  const int row0 = ch0 >> 2, c80 = (ch0 & 3) * 8;
  const int row1 = ch1 >> 2, c81 = (ch1 & 3) * 8;
  const unsigned short* Ap0 = A  + (size_t)(bm + row0) * K + c80;
  const unsigned short* Ap1 = A  + (size_t)(bm + row1) * K + c81;
  const unsigned short* Bp0 = Bt + (size_t)(bn + row0) * K + c80;
  const unsigned short* Bp1 = Bt + (size_t)(bn + row1) * K + c81;

  v8f acc[4][2];
#pragma unroll
  for (int i = 0; i < 4; ++i)
#pragma unroll
    for (int j = 0; j < 2; ++j)
#pragma unroll
      for (int e = 0; e < 8; ++e) acc[i][j][e] = 0.f;

  union Frag { uint4 q[2]; v16bf v; };

  uint4 ra0, ra1, rb0, rb1;                       // register staging for next tile

  const int niter = K / 32;

  // prologue: tile 0 -> LDS[0]; tile 1 -> registers
  ra0 = *(const uint4*)(Ap0); ra1 = *(const uint4*)(Ap1);
  rb0 = *(const uint4*)(Bp0); rb1 = *(const uint4*)(Bp1);
  *(uint4*)&As[0][row0 * LDT + c80] = ra0;
  *(uint4*)&As[0][row1 * LDT + c81] = ra1;
  *(uint4*)&Bs[0][row0 * LDT + c80] = rb0;
  *(uint4*)&Bs[0][row1 * LDT + c81] = rb1;
  if (niter > 1) {
    ra0 = *(const uint4*)(Ap0 + 32); ra1 = *(const uint4*)(Ap1 + 32);
    rb0 = *(const uint4*)(Bp0 + 32); rb1 = *(const uint4*)(Bp1 + 32);
  }
  __syncthreads();

  for (int i = 0; i < niter; ++i) {
    const int cur = i & 1;

    // stage tile i+1 (in registers) into the other LDS buffer; the buffer it
    // overwrites was last read in iteration i-1, protected by that barrier.
    if (i + 1 < niter) {
      const int nxt = cur ^ 1;
      *(uint4*)&As[nxt][row0 * LDT + c80] = ra0;
      *(uint4*)&As[nxt][row1 * LDT + c81] = ra1;
      *(uint4*)&Bs[nxt][row0 * LDT + c80] = rb0;
      *(uint4*)&Bs[nxt][row1 * LDT + c81] = rb1;
      if (i + 2 < niter) {
        const int k2 = (i + 2) * 32;
        ra0 = *(const uint4*)(Ap0 + k2); ra1 = *(const uint4*)(Ap1 + k2);
        rb0 = *(const uint4*)(Bp0 + k2); rb1 = *(const uint4*)(Bp1 + k2);
      }
      if (i + 3 < niter) {
        const int k3 = (i + 3) * 32;
        __builtin_prefetch(Ap0 + k3, 0, 1);       // global_prefetch_b8
        __builtin_prefetch(Bp0 + k3, 0, 1);
      }
    }

    // A fragment: lane holds K = {kb..kb+7} U {kb+16..kb+23}, kb = 8*half
    v16bf af[4];
#pragma unroll
    for (int mt = 0; mt < 4; ++mt) {
      int m = wm * 64 + mt * 16 + lr;
      Frag f;
      f.q[0] = *(const uint4*)&As[cur][m * LDT + half * 8];
      f.q[1] = *(const uint4*)&As[cur][m * LDT + half * 8 + 16];
      af[mt] = f.v;
    }
    // B fragment: lane holds K = {kb..kb+7} U {kb+16..kb+23}, kb = 8*half
    v16bf bf[2];
#pragma unroll
    for (int nt = 0; nt < 2; ++nt) {
      int n = wn * 32 + nt * 16 + lr;
      Frag f;
      f.q[0] = *(const uint4*)&Bs[cur][n * LDT + half * 8];
      f.q[1] = *(const uint4*)&Bs[cur][n * LDT + half * 8 + 16];
      bf[nt] = f.v;
    }
#pragma unroll
    for (int mt = 0; mt < 4; ++mt)
#pragma unroll
      for (int nt = 0; nt < 2; ++nt)
        acc[mt][nt] = __builtin_amdgcn_wmma_f32_16x16x32_bf16(
            false, af[mt], false, bf[nt], (short)0, acc[mt][nt], false, false);

    __syncthreads();
  }

  // epilogue: D layout -> m = r + 8*half, n = lr
#pragma unroll
  for (int mt = 0; mt < 4; ++mt) {
#pragma unroll
    for (int nt = 0; nt < 2; ++nt) {
      int n  = bn + wn * 32 + nt * 16 + lr;
      float bv = bias[n];
#pragma unroll
      for (int r = 0; r < 8; ++r) {
        int m   = bm + wm * 64 + mt * 16 + r + half * 8;
        float v = acc[mt][nt][r] + bv;
        if (ACT == 0) {
          v = v > 0.f ? v : 0.f;
          ((unsigned short*)Cout)[(size_t)m * Ndim + n] = f2bf(v);
        } else if (ACT == 1) {
          ((unsigned short*)Cout)[(size_t)m * Ndim + n] = f2bf(tanhf(v));
        } else {
          ((float*)Cout)[(size_t)m * Ndim + n] = v;
        }
      }
    }
  }
}

// ---------------- host launch ----------------
extern "C" void kernel_launch(void* const* d_in, const int* in_sizes, int n_in,
                              void* d_out, int out_size, void* d_ws, size_t ws_size,
                              hipStream_t stream) {
  (void)in_sizes; (void)n_in; (void)out_size; (void)ws_size;
  const float* x     = (const float*)d_in[0];
  const float* w     = (const float*)d_in[1];
  const float* emb_w = (const float*)d_in[2];
  const float* emb_b = (const float*)d_in[3];
  const float* lin_w = (const float*)d_in[4];
  const float* lin_b = (const float*)d_in[5];
  const float* out_w = (const float*)d_in[6];
  const float* out_b = (const float*)d_in[7];

  char* ws = (char*)d_ws;
  unsigned short* xs    = (unsigned short*)(ws + OFF_XS);
  unsigned short* embT  = (unsigned short*)(ws + OFF_EMBT);
  unsigned short* linT0 = (unsigned short*)(ws + OFF_LINT0);
  unsigned short* linT1 = (unsigned short*)(ws + OFF_LINT1);
  unsigned short* outT  = (unsigned short*)(ws + OFF_OUTT);
  float*          ebias = (float*)(ws + OFF_EBIAS);
  unsigned short* h0    = (unsigned short*)(ws + OFF_H0);
  unsigned short* h1    = (unsigned short*)(ws + OFF_H1);

  // bias prep + gate-scaled bf16 activations
  ebias_kernel<<<EMB_ / 256, 256, 0, stream>>>(emb_b, ebias);
  gates_kernel<<<M_, 256, 0, stream>>>(x, w, xs);

  // weight convert + transpose: [K][N] fp32 -> [N][K] bf16
  convT_kernel<<<dim3(EMB_ / 32, D_ / 32),   dim3(32, 8), 0, stream>>>(emb_w, embT, D_, EMB_);
  convT_kernel<<<dim3(EMB_ / 32, EMB_ / 32), dim3(32, 8), 0, stream>>>(lin_w, linT0, EMB_, EMB_);
  convT_kernel<<<dim3(EMB_ / 32, EMB_ / 32), dim3(32, 8), 0, stream>>>(lin_w + (size_t)EMB_ * EMB_,
                                                                       linT1, EMB_, EMB_);
  convT_kernel<<<dim3(OUT_ / 32, EMB_ / 32), dim3(32, 8), 0, stream>>>(out_w, outT, EMB_, OUT_);

  // GEMM chain
  gemm_kernel<0><<<dim3(EMB_ / 128, M_ / 128), 256, 0, stream>>>(xs, embT, ebias, h0, EMB_, D_);
  gemm_kernel<1><<<dim3(EMB_ / 128, M_ / 128), 256, 0, stream>>>(h0, linT0, lin_b, h1, EMB_, EMB_);
  gemm_kernel<1><<<dim3(EMB_ / 128, M_ / 128), 256, 0, stream>>>(h1, linT1, lin_b + EMB_, h0, EMB_, EMB_);
  gemm_kernel<2><<<dim3(OUT_ / 128, M_ / 128), 256, 0, stream>>>(h0, outT, out_b, d_out, OUT_, EMB_);
}
